// AttentionLSTM_Dot_28192165331638
// MI455X (gfx1250) — matrix operand order repaired and checked by reading, not compiled, not run on hardware
//
#include <hip/hip_runtime.h>
#include <hip/hip_bf16.h>

// ---------------------------------------------------------------------------
// AttentionLSTM on gfx1250 (CDNA5, wave32, WMMA 16x16x32 bf16 / f32 accum)
// B=64, T=512, V=32000, E=512, H=512, 4H=2048, O=1
// ---------------------------------------------------------------------------

#define BB   64
#define TT   512
#define EE   512
#define HH   512
#define G4   2048          // 4*H
#define BT   (BB*TT)       // 32768 rows of the big GEMM

typedef __bf16 bf16_t;
typedef bf16_t bf16x8  __attribute__((ext_vector_type(8)));
typedef bf16_t bf16x16 __attribute__((ext_vector_type(16)));
typedef float  f32x8   __attribute__((ext_vector_type(8)));

// ---- WMMA helpers ----------------------------------------------------------

__device__ __forceinline__ f32x8 wmma_bf16(bf16x16 a, bf16x16 b, f32x8 c) {
  // v_wmma_f32_16x16x32_bf16  (neg_a, A, neg_b, B, c_mod, C, reuse_a, reuse_b)
  return __builtin_amdgcn_wmma_f32_16x16x32_bf16(false, a, false, b,
                                                 (short)0, c, false, false);
}

__device__ __forceinline__ bf16x8 ld8(const bf16_t* p) {
  return *reinterpret_cast<const bf16x8*>(p);
}

// A fragment 16x32 (MxK), bf16, row-major source (ld elements per row).
// Lane l (l&15 = row, l>>4 = half): K = {8h..8h+7} then {16+8h..23+8h}.
__device__ __forceinline__ bf16x16 load_a(const bf16_t* base, int ld, int lane) {
  int row = lane & 15, half = lane >> 4;
  const bf16_t* p = base + (size_t)row * ld + 8 * half;
  union { bf16x16 v; bf16x8 h[2]; } u;
  u.h[0] = ld8(p);
  u.h[1] = ld8(p + 16);
  return u.v;
}

// B fragment 32x16 (KxN) taken from a row-major [N x K] (pre-transposed)
// matrix. Lane l (l&15 = col n, l>>4 = half): K = {8h..8h+7} then {16+8h..23+8h}.
__device__ __forceinline__ bf16x16 load_b(const bf16_t* base, int ld, int lane) {
  int col = lane & 15, half = lane >> 4;
  const bf16_t* p = base + (size_t)col * ld + 8 * half;
  union { bf16x16 v; bf16x8 h[2]; } u;
  u.h[0] = ld8(p);
  u.h[1] = ld8(p + 16);
  return u.v;
}

__device__ __forceinline__ float sigm(float x) {
  return 1.0f / (1.0f + __expf(-x));
}

// ---- Kernel 1: transpose+convert Wi [E,4H] and Wh [H,4H] -> bf16 [4H,E] ----

__global__ __launch_bounds__(256) void k_prep_w(const float* __restrict__ Wi,
                                                const float* __restrict__ Wh,
                                                bf16_t* __restrict__ WiT,
                                                bf16_t* __restrict__ WhT) {
  int idx = blockIdx.x * 256 + threadIdx.x;            // 0 .. 2*4H*E-1
  int which = idx >= (G4 * EE);
  int inner = idx - which * (G4 * EE);
  int g = inner >> 9;                                  // /E
  int e = inner & (EE - 1);
  const float* src = which ? Wh : Wi;
  bf16_t* dst = which ? WhT : WiT;
  dst[(size_t)g * EE + e] = (bf16_t)src[(size_t)e * G4 + g];
}

// ---- Kernel 2: gather embedding rows into bf16 x[t*B+b][e] -----------------

__global__ __launch_bounds__(256) void k_gather(const int* __restrict__ words,
                                                const float* __restrict__ emb,
                                                bf16_t* __restrict__ xbf) {
  size_t idx = (size_t)blockIdx.x * 256 + threadIdx.x; // B*T*E threads
  int e = (int)(idx & (EE - 1));
  int r = (int)(idx >> 9);                             // t*B + b
  int b = r & (BB - 1);
  int t = r >> 6;                                      // /B
  int w = words[b * TT + t];
  xbf[idx] = (bf16_t)emb[(size_t)w * EE + e];
}

// ---- Kernel 3: big GEMM  xp[r][g] = x[r][:] . WiT[g][:] + bi[g] ------------
// Wave tile 32(M) x 64(N); 8 accum frags; K loop over 512 in steps of 32.

__global__ __launch_bounds__(256) void k_gemm_xp(const bf16_t* __restrict__ x,
                                                 const bf16_t* __restrict__ WiT,
                                                 const float* __restrict__ bi,
                                                 float* __restrict__ xp) {
  int lane = threadIdx.x & 31;
  int wid  = blockIdx.x * 8 + (threadIdx.x >> 5);      // 0 .. 32767
  int nt = wid & 31;                                   // 32 N tiles of 64
  int mt = wid >> 5;                                   // 1024 M tiles of 32
  int Mb = mt * 32;
  int Nb = nt * 64;

  f32x8 acc[2][4] = {};
  for (int k = 0; k < EE; k += 32) {
    bf16x16 a0 = load_a(x + (size_t)Mb * EE + k, EE, lane);
    bf16x16 a1 = load_a(x + (size_t)(Mb + 16) * EE + k, EE, lane);
#pragma unroll
    for (int j = 0; j < 4; ++j) {
      bf16x16 bj = load_b(WiT + (size_t)(Nb + 16 * j) * EE + k, EE, lane);
      acc[0][j] = wmma_bf16(a0, bj, acc[0][j]);
      acc[1][j] = wmma_bf16(a1, bj, acc[1][j]);
    }
  }

  int n = lane & 15, half = lane >> 4;
#pragma unroll
  for (int i = 0; i < 2; ++i) {
#pragma unroll
    for (int j = 0; j < 4; ++j) {
      int col = Nb + 16 * j + n;
      float bias = bi[col];
#pragma unroll
      for (int r = 0; r < 8; ++r) {
        int row = Mb + 16 * i + r + 8 * half;          // row = t*B+b
        xp[(size_t)row * G4 + col] = acc[i][j][r] + bias;
      }
    }
  }
}

// ---- Kernel 4: persistent LSTM scan ----------------------------------------
// 16 blocks x 256 threads = 128 waves. Wave w: hidden cols [16*(w>>2)..+16),
// batch rows [16*(w&3)..+16), ALL four gates -> c state stays in registers.
// One monotonic atomic grid barrier per timestep.

__global__ __launch_bounds__(256) void k_lstm(const float* __restrict__ xp,
                                              const bf16_t* __restrict__ WhT,
                                              const float* __restrict__ bh,
                                              bf16_t* __restrict__ hbf,
                                              float* __restrict__ hs,
                                              float* __restrict__ dout,
                                              unsigned* __restrict__ bar) {
  int lane = threadIdx.x & 31;
  int w = blockIdx.x * 8 + (threadIdx.x >> 5);         // 0..127
  int cb = (w >> 2) * 16;                              // hidden col base
  int rb = (w & 3) * 16;                               // batch row base
  int n = lane & 15, half = lane >> 4;
  int col = cb + n;

  float bhv[4];
#pragma unroll
  for (int q = 0; q < 4; ++q) bhv[q] = bh[q * HH + col];

  f32x8 c = {};
  for (int t = 0; t < TT; ++t) {
    f32x8 acc[4] = {};
    if (t > 0) {
      for (int k = 0; k < HH; k += 32) {
        bf16x16 a = load_a(hbf + (size_t)rb * HH + k, HH, lane);
#pragma unroll
        for (int q = 0; q < 4; ++q) {
          bf16x16 bq = load_b(WhT + (size_t)(q * HH + cb) * EE + k, EE, lane);
          acc[q] = wmma_bf16(a, bq, acc[q]);
        }
      }
    }
    const float* xpt = xp + (size_t)t * (BB * G4);
#pragma unroll
    for (int r = 0; r < 8; ++r) {
      int row = rb + r + 8 * half;                     // batch index
      const float* xr = xpt + (size_t)row * G4;
      float gr = xr[col]             + acc[0][r] + bhv[0];
      float gf = xr[HH + col]        + acc[1][r] + bhv[1];
      float gg = xr[2 * HH + col]    + acc[2][r] + bhv[2];
      float go = xr[3 * HH + col]    + acc[3][r] + bhv[3];
      float rt = sigm(gr), ft = sigm(gf);
      float gt = tanhf(gg), ot = sigm(go);
      float cv = ft * c[r] + rt * gt;
      c[r] = cv;
      float ht = ot * tanhf(cv);
      hs[(size_t)t * (BB * HH) + (size_t)row * HH + col] = ht;
      hbf[(size_t)row * HH + col] = (bf16_t)ht;
      if (t == TT - 1) dout[BB + (size_t)row * HH + col] = ht;  // final h
    }
    // device-wide barrier: all 16 blocks must finish writing h before next t
    __threadfence();
    __syncthreads();
    if (threadIdx.x == 0) {
      atomicAdd(bar, 1u);
      unsigned target = 16u * (unsigned)(t + 1);
      while (atomicAdd(bar, 0u) < target) __builtin_amdgcn_s_sleep(2);
    }
    __syncthreads();
    __threadfence();
  }
}

// ---- Kernel 5: final_hidden = h @ W_lstm + b_lstm   [64x512] ---------------

__global__ __launch_bounds__(256) void k_fh(const float* __restrict__ hs,
                                            const float* __restrict__ W_lstm,
                                            const float* __restrict__ b_lstm,
                                            float* __restrict__ fh) {
  int idx = blockIdx.x * 256 + threadIdx.x;            // 32768
  int b = idx >> 9, jc = idx & (HH - 1);
  const float* hrow = hs + (size_t)(TT - 1) * (BB * HH) + (size_t)b * HH;
  float a = b_lstm[jc];
  for (int k = 0; k < HH; ++k) a += hrow[k] * W_lstm[(size_t)k * HH + jc];
  fh[idx] = a;
}

// ---- Kernel 6: fused score/softmax/att/output (one block per batch row) ----

__global__ __launch_bounds__(256) void k_attn(const float* __restrict__ hs,
                                              const float* __restrict__ fh,
                                              const float* __restrict__ W_att,
                                              const float* __restrict__ b_att,
                                              float* __restrict__ out) {
  __shared__ float sc[TT];
  __shared__ float red[256];
  int b = blockIdx.x, tid = threadIdx.x;
  const float* frow = fh + (size_t)b * HH;

  // score[b][t] = hs[t][b][:] . fh[b][:]
  for (int t = tid; t < TT; t += 256) {
    const float4* h4 = (const float4*)(hs + (size_t)t * (BB * HH) + (size_t)b * HH);
    const float4* f4 = (const float4*)frow;
    float d = 0.f;
    for (int i = 0; i < HH / 4; ++i) {
      float4 x = h4[i], y = f4[i];
      d += x.x * y.x + x.y * y.y + x.z * y.z + x.w * y.w;
    }
    sc[t] = d;
  }
  __syncthreads();

  // softmax over t
  float lm = fmaxf(sc[tid], sc[tid + 256]);
  red[tid] = lm;
  __syncthreads();
  for (int s = 128; s > 0; s >>= 1) {
    if (tid < s) red[tid] = fmaxf(red[tid], red[tid + s]);
    __syncthreads();
  }
  float mx = red[0];
  __syncthreads();
  float ls = 0.f;
  for (int t = tid; t < TT; t += 256) {
    float e = __expf(sc[t] - mx);
    sc[t] = e;
    ls += e;
  }
  red[tid] = ls;
  __syncthreads();
  for (int s = 128; s > 0; s >>= 1) {
    if (tid < s) red[tid] += red[tid + s];
    __syncthreads();
  }
  float inv = 1.0f / red[0];
  __syncthreads();

  // att[b][h] = sum_t hs[t][b][h]*dist[t]; fuse out = [fh,att] @ W_att
  float part = 0.f;
  for (int h = tid; h < HH; h += 256) {
    float a = 0.f;
    for (int t = 0; t < TT; ++t)
      a += hs[(size_t)t * (BB * HH) + (size_t)b * HH + h] * sc[t];
    a *= inv;
    part += frow[h] * W_att[h] + a * W_att[HH + h];
  }
  red[tid] = part;
  __syncthreads();
  for (int s = 128; s > 0; s >>= 1) {
    if (tid < s) red[tid] += red[tid + s];
    __syncthreads();
  }
  if (tid == 0) out[b] = sigm(red[0] + b_att[0]);
}

// ---------------------------------------------------------------------------

extern "C" void kernel_launch(void* const* d_in, const int* in_sizes, int n_in,
                              void* d_out, int out_size, void* d_ws, size_t ws_size,
                              hipStream_t stream) {
  const int*   words  = (const int*)d_in[0];
  const float* emb    = (const float*)d_in[1];
  const float* Wi     = (const float*)d_in[2];
  const float* bi     = (const float*)d_in[3];
  const float* Wh     = (const float*)d_in[4];
  const float* bh     = (const float*)d_in[5];
  const float* W_lstm = (const float*)d_in[6];
  const float* b_lstm = (const float*)d_in[7];
  const float* W_att  = (const float*)d_in[8];
  const float* b_att  = (const float*)d_in[9];
  float* out = (float*)d_out;

  char* ws = (char*)d_ws;
  size_t off = 0;
  auto take = [&](size_t bytes) { size_t o = off; off += (bytes + 255) & ~(size_t)255; return o; };
  size_t o_bar = take(256);
  size_t o_wit = take((size_t)G4 * EE * 2);            // 2 MB
  size_t o_wht = take((size_t)G4 * EE * 2);            // 2 MB
  size_t o_x   = take((size_t)BT * EE * 2);            // 33.5 MB
  size_t o_xp  = take((size_t)BT * G4 * 4);            // 268 MB
  size_t o_hs  = take((size_t)TT * BB * HH * 4);       // 67 MB
  size_t o_fh  = take((size_t)BB * HH * 4);
  size_t o_hb  = take((size_t)BB * HH * 2);

  unsigned* bar  = (unsigned*)(ws + o_bar);
  bf16_t* WiT    = (bf16_t*)(ws + o_wit);
  bf16_t* WhT    = (bf16_t*)(ws + o_wht);
  bf16_t* xbf    = (bf16_t*)(ws + o_x);
  float*  xp     = (float*)(ws + o_xp);
  float*  hs     = (float*)(ws + o_hs);
  float*  fh     = (float*)(ws + o_fh);
  bf16_t* hbf    = (bf16_t*)(ws + o_hb);

  hipMemsetAsync(ws + o_bar, 0, 256, stream);

  k_prep_w<<<(2 * G4 * EE) / 256, 256, 0, stream>>>(Wi, Wh, WiT, WhT);
  k_gather<<<(BT * EE) / 256, 256, 0, stream>>>(words, emb, xbf);
  k_gemm_xp<<<(BT / 32) * (G4 / 64) / 8, 256, 0, stream>>>(xbf, WiT, bi, xp);
  k_lstm<<<16, 256, 0, stream>>>(xp, WhT, bh, hbf, hs, out, bar);
  k_fh<<<(BB * HH) / 256, 256, 0, stream>>>(hs, W_lstm, b_lstm, fh);
  k_attn<<<BB, 256, 0, stream>>>(hs, fh, W_att, b_att, out);
}
